// MultiHeadAttention_15668040696302
// MI455X (gfx1250) — matrix operand order repaired and checked by reading, not compiled, not run on hardware
//
#include <hip/hip_runtime.h>
#include <hip/hip_bf16.h>

// ---------------------------------------------------------------------------
// MI455X (gfx1250, wave32) multi-head attention, bf16 WMMA everywhere.
// D = A(16x32 bf16) x B(32x16 bf16) + C(16x16 f32) via v_wmma_f32_16x16x32_bf16
// Async global->LDS staging (ASYNCcnt) where no conversion is needed.
// ---------------------------------------------------------------------------

typedef __bf16 bf16;
typedef __bf16 v16bf __attribute__((ext_vector_type(16)));
typedef __bf16 v8bf  __attribute__((ext_vector_type(8)));
typedef float  v8f   __attribute__((ext_vector_type(8)));
typedef float  v4f   __attribute__((ext_vector_type(4)));
typedef int    v4i   __attribute__((ext_vector_type(4)));

#define DEV __device__ __forceinline__

DEV bf16 f2bf(float x) { return (bf16)x; }

// ---------------- async global->LDS (CDNA5 ASYNCcnt path) -------------------
#if defined(__AMDGCN__) && __has_builtin(__builtin_amdgcn_global_load_async_to_lds_b128)
#define HAVE_ASYNC_LDS 1
typedef __attribute__((address_space(1))) v4i* gv4_ptr;  // global int4*
typedef __attribute__((address_space(3))) v4i* lv4_ptr;  // LDS int4*
#else
#define HAVE_ASYNC_LDS 0
#endif

// copy 16 bytes (8 bf16) global -> LDS
DEV void async_copy16(const bf16* g, bf16* l) {
#if HAVE_ASYNC_LDS
  gv4_ptr gp = (gv4_ptr)(unsigned long long)(const void*)g;
  lv4_ptr lp = (lv4_ptr)(unsigned int)(unsigned long long)(void*)l;
  __builtin_amdgcn_global_load_async_to_lds_b128(gp, lp, 0, 0);
#else
  *(v8bf*)l = *(const v8bf*)g;
#endif
}

DEV void async_wait0() {
#if HAVE_ASYNC_LDS
#if __has_builtin(__builtin_amdgcn_s_wait_asynccnt)
  __builtin_amdgcn_s_wait_asynccnt(0);
#else
  asm volatile("s_wait_asynccnt 0x0" ::: "memory");
#endif
#endif
}

// ---------------- DPP16 butterfly reductions (pure VALU, no LDS) ------------
template <int CTRL>
DEV float dpp_f(float x) {
  int i = __builtin_bit_cast(int, x);
  return __builtin_bit_cast(
      float, __builtin_amdgcn_update_dpp(i, i, CTRL, 0xF, 0xF, false));
}
// reduce across the 16-lane half-wave (C-matrix row lives in 16 lanes)
DEV float red_max16(float x) {
  x = fmaxf(x, dpp_f<0xB1>(x));   // quad_perm [1,0,3,2]  (xor 1)
  x = fmaxf(x, dpp_f<0x4E>(x));   // quad_perm [2,3,0,1]  (xor 2)
  x = fmaxf(x, dpp_f<0x141>(x));  // row_half_mirror      (covers xor 4)
  x = fmaxf(x, dpp_f<0x140>(x));  // row_mirror           (covers xor 8)
  return x;
}
DEV float red_sum16(float x) {
  x += dpp_f<0xB1>(x);
  x += dpp_f<0x4E>(x);
  x += dpp_f<0x141>(x);
  x += dpp_f<0x140>(x);
  return x;
}

// ---------------- WMMA fragment helpers (ISA 7.12.2 layouts) ----------------
DEV v16bf make_frag(v8bf lo, v8bf hi) {
  v16bf r;
#pragma unroll
  for (int i = 0; i < 8; ++i) { r[i] = lo[i]; r[i + 8] = hi[i]; }
  return r;
}

// A fragment (16x32 MxK): lanes 0-15 hold K={0..7,16..23}, lanes 16-31 hold
// K={8..15,24..31}.  hl = lane/16, kwin selects the 32-wide K window.
DEV v16bf frag_A(const bf16* rowBase, int kwin, int hl) {
  v8bf lo = *(const v8bf*)(rowBase + kwin + hl * 8);
  v8bf hi = *(const v8bf*)(rowBase + kwin + 16 + hl * 8);
  return make_frag(lo, hi);
}

// B fragment (32x16 KxN) from LDS staged [N][K]: lanes 0-15 col=lane K={0..7,16..23},
// lanes 16-31 K={8..15,24..31}.
DEV v16bf frag_B(const bf16* colBase, int kwin, int hl) {
  v8bf lo = *(const v8bf*)(colBase + kwin + hl * 8);
  v8bf hi = *(const v8bf*)(colBase + kwin + 16 + hl * 8);
  return make_frag(lo, hi);
}

DEV v8f wmma_bf16(v16bf a, v16bf b, v8f c) {
  return __builtin_amdgcn_wmma_f32_16x16x32_bf16(
      false, a, false, b, (short)0, c, false, false);
}

// ---------------------------------------------------------------------------
// GEMM:  C[M,N] = A[M,K] @ W[K,N] + bias[N]     (M=4096, N=K=1024)
// Block tile 128x64, K-step 64 (8 WMMAs per barrier pair); 8 waves of 32x32.
// ---------------------------------------------------------------------------
constexpr int GM = 4096, GN = 1024, GK = 1024;

template <typename InT, typename OutT>
__global__ __launch_bounds__(256)
void gemm_bias_kernel(const InT* __restrict__ A, const float* __restrict__ W,
                      const float* __restrict__ bias, OutT* __restrict__ C) {
  __shared__ __attribute__((aligned(16))) bf16 As[128][64];
  __shared__ __attribute__((aligned(16))) bf16 Bt[64][64];  // [N][K]

  const int tid  = threadIdx.x;
  const int wave = tid >> 5;
  const int lane = tid & 31;
  const int hl   = lane >> 4;
  const int l16  = lane & 15;
  const int wm   = wave >> 1;   // 0..3
  const int wn   = wave & 1;    // 0..1
  const int m0   = blockIdx.y * 128;
  const int n0   = blockIdx.x * 64;

  v8f acc[2][2] = {};

  const int arow = tid >> 1;          // 0..127
  const int aseg = (tid & 1) * 32;    // 0 / 32
  const int bk   = tid >> 2;          // 0..63
  const int bn   = (tid & 3) * 16;    // 0..48

  for (int k0 = 0; k0 < GK; k0 += 64) {
    // ---- stage A tile (fp32 converts through VGPRs; bf16 goes async)
    {
      const InT* src = A + (size_t)(m0 + arow) * GK + k0 + aseg;
      bf16* dst = &As[arow][aseg];
      if constexpr (__is_same(InT, float)) {
        const v4f* s4 = (const v4f*)src;
#pragma unroll
        for (int i = 0; i < 4; ++i) {
          v4f f0 = s4[2 * i], f1 = s4[2 * i + 1];
          v8bf h;
#pragma unroll
          for (int j = 0; j < 4; ++j) { h[j] = f2bf(f0[j]); h[j + 4] = f2bf(f1[j]); }
          *(v8bf*)(dst + 8 * i) = h;
        }
      } else {
#pragma unroll
        for (int i = 0; i < 4; ++i) async_copy16(src + 8 * i, dst + 8 * i);
      }
    }
    // ---- stage W tile transposed into [N][K]
    {
      const float* ws = W + (size_t)(k0 + bk) * GN + n0 + bn;
      const v4f* s4 = (const v4f*)ws;
#pragma unroll
      for (int i = 0; i < 4; ++i) {
        v4f f = s4[i];
#pragma unroll
        for (int j = 0; j < 4; ++j) Bt[bn + 4 * i + j][bk] = f2bf(f[j]);
      }
    }
    async_wait0();
    __syncthreads();

    v16bf af[2][2], bfr[2][2];
#pragma unroll
    for (int tm = 0; tm < 2; ++tm)
#pragma unroll
      for (int s = 0; s < 2; ++s)
        af[tm][s] = frag_A(&As[wm * 32 + tm * 16 + l16][0], s * 32, hl);
#pragma unroll
    for (int tn = 0; tn < 2; ++tn)
#pragma unroll
      for (int s = 0; s < 2; ++s)
        bfr[tn][s] = frag_B(&Bt[wn * 32 + tn * 16 + l16][0], s * 32, hl);
#pragma unroll
    for (int tm = 0; tm < 2; ++tm)
#pragma unroll
      for (int tn = 0; tn < 2; ++tn)
#pragma unroll
        for (int s = 0; s < 2; ++s)
          acc[tm][tn] = wmma_bf16(af[tm][s], bfr[tn][s], acc[tm][tn]);
    __syncthreads();
  }

  // epilogue: bias + store.  C layout: vgpr r -> row r + 8*hl, col = l16.
#pragma unroll
  for (int tm = 0; tm < 2; ++tm) {
#pragma unroll
    for (int tn = 0; tn < 2; ++tn) {
      const int col  = n0 + wn * 32 + tn * 16 + l16;
      const float bc = bias[col];
      const int rb   = m0 + wm * 32 + tm * 16 + hl * 8;
#pragma unroll
      for (int r = 0; r < 8; ++r) {
        float val = acc[tm][tn][r] + bc;
        if constexpr (__is_same(OutT, float))
          C[(size_t)(rb + r) * GN + col] = val;
        else
          C[(size_t)(rb + r) * GN + col] = f2bf(val);
      }
    }
  }
}

// ---------------------------------------------------------------------------
// Flash attention: one block per (b, h, 128 query rows); 8 waves x 16 rows.
// Online softmax in fp32 (DPP reductions), QK^T and PV on bf16 WMMA.
// ---------------------------------------------------------------------------
constexpr int AB = 2, ASEQ = 2048, ADM = 1024, AH = 16, ADEP = 64;

__global__ __launch_bounds__(256)
void mha_attention_kernel(const bf16* __restrict__ Q, const bf16* __restrict__ K,
                          const bf16* __restrict__ V, bf16* __restrict__ O) {
  __shared__ __attribute__((aligned(16))) bf16 Qs[128][64];
  __shared__ __attribute__((aligned(16))) bf16 Ks[64][64];      // [key][chan]
  __shared__ __attribute__((aligned(16))) bf16 Vt[64][64];      // [chan][key]
  __shared__ __attribute__((aligned(16))) bf16 Pw[8][16][64];   // per-wave P

  const int tid  = threadIdx.x;
  const int wave = tid >> 5;
  const int lane = tid & 31;
  const int hl   = lane >> 4;
  const int l16  = lane & 15;

  const int q0 = blockIdx.x * 128;
  const int h  = blockIdx.y;
  const int b  = blockIdx.z;
  const size_t headOff = (size_t)h * ADEP;
  const size_t batOff  = (size_t)b * ASEQ * ADM;

  // stage this block's 128x64 Q slice (async: already bf16)
  {
    const int r = tid >> 1;
    const int c = (tid & 1) * 32;
    const bf16* g = Q + batOff + (size_t)(q0 + r) * ADM + headOff + c;
    bf16* l = &Qs[r][c];
#pragma unroll
    for (int i = 0; i < 4; ++i) async_copy16(g + 8 * i, l + 8 * i);
  }
  async_wait0();
  __syncthreads();

  v16bf qf[2];
  qf[0] = frag_A(&Qs[wave * 16 + l16][0], 0, hl);
  qf[1] = frag_A(&Qs[wave * 16 + l16][0], 32, hl);

  v8f o[4] = {};
  float mrun[8], lrun[8];
#pragma unroll
  for (int r = 0; r < 8; ++r) { mrun[r] = -__builtin_inff(); lrun[r] = 0.f; }

  constexpr float SC = 0.125f * 1.44269504088896f;  // 1/sqrt(64) * log2(e)

  const int sr  = tid >> 2;         // key row 0..63
  const int scg = (tid & 3) * 16;   // chan group

  for (int j = 0; j < ASEQ; j += 64) {
    __syncthreads();  // previous block's Ks/Vt fully consumed
    // stage K block [key][chan] (async bf16 copy; B-operand layout for QK^T)
    {
      const bf16* g = K + batOff + (size_t)(j + sr) * ADM + headOff + scg;
      async_copy16(g, &Ks[sr][scg]);
      async_copy16(g + 8, &Ks[sr][scg + 8]);
    }
    // stage V transposed [chan][key] (B-operand layout for P@V)
    {
      const v8bf* s = (const v8bf*)(V + batOff + (size_t)(j + sr) * ADM + headOff + scg);
      v8bf x0 = s[0], x1 = s[1];
#pragma unroll
      for (int t = 0; t < 8; ++t) {
        Vt[scg + t][sr]     = x0[t];
        Vt[scg + 8 + t][sr] = x1[t];
      }
    }
    async_wait0();
    __syncthreads();

    // logits: 4 tiles of 16x16 over 64 keys, K-dim = depth 64 (2 WMMAs each)
    v8f c4[4] = {};
#pragma unroll
    for (int t = 0; t < 4; ++t) {
      v16bf kb0 = frag_B(&Ks[t * 16 + l16][0], 0, hl);
      v16bf kb1 = frag_B(&Ks[t * 16 + l16][0], 32, hl);
      c4[t] = wmma_bf16(qf[0], kb0, c4[t]);
      c4[t] = wmma_bf16(qf[1], kb1, c4[t]);
    }

    // online softmax: row = r + 8*hl, row values spread over the 16-lane half
    float mnew[8], alpha[8];
#pragma unroll
    for (int r = 0; r < 8; ++r) {
      float mx = fmaxf(fmaxf(c4[0][r], c4[1][r]), fmaxf(c4[2][r], c4[3][r]));
      mx = red_max16(mx);
      mnew[r]  = fmaxf(mrun[r], mx);
      alpha[r] = exp2f(SC * (mrun[r] - mnew[r]));
      mrun[r]  = mnew[r];
    }
    float rs[8];
#pragma unroll
    for (int r = 0; r < 8; ++r) rs[r] = 0.f;
#pragma unroll
    for (int t = 0; t < 4; ++t) {
#pragma unroll
      for (int r = 0; r < 8; ++r) {
        float p = exp2f(SC * (c4[t][r] - mnew[r]));
        rs[r] += p;
        Pw[wave][r + hl * 8][t * 16 + l16] = f2bf(p);  // D-layout -> row major
      }
    }
#pragma unroll
    for (int r = 0; r < 8; ++r) {
      lrun[r] = lrun[r] * alpha[r] + red_sum16(rs[r]);
    }
#pragma unroll
    for (int t = 0; t < 4; ++t)
#pragma unroll
      for (int r = 0; r < 8; ++r) o[t][r] *= alpha[r];

    // O += P @ V   (A = P from per-wave LDS, B = Vt)
    v16bf pf[2];
    pf[0] = frag_A(&Pw[wave][l16][0], 0, hl);
    pf[1] = frag_A(&Pw[wave][l16][0], 32, hl);
#pragma unroll
    for (int t = 0; t < 4; ++t) {
      v16bf vb0 = frag_B(&Vt[t * 16 + l16][0], 0, hl);
      v16bf vb1 = frag_B(&Vt[t * 16 + l16][0], 32, hl);
      o[t] = wmma_bf16(pf[0], vb0, o[t]);
      o[t] = wmma_bf16(pf[1], vb1, o[t]);
    }
  }

  // normalize and store bf16 attention output [B,S,D] (heads concatenated)
#pragma unroll
  for (int r = 0; r < 8; ++r) lrun[r] = 1.f / lrun[r];
#pragma unroll
  for (int t = 0; t < 4; ++t) {
    const size_t col = headOff + t * 16 + l16;
#pragma unroll
    for (int r = 0; r < 8; ++r) {
      const int row = q0 + wave * 16 + r + hl * 8;
      O[batOff + (size_t)row * ADM + col] = f2bf(o[t][r] * lrun[r]);
    }
  }
}

// ---------------------------------------------------------------------------
// Host-side launch
// ---------------------------------------------------------------------------
extern "C" void kernel_launch(void* const* d_in, const int* in_sizes, int n_in,
                              void* d_out, int out_size, void* d_ws, size_t ws_size,
                              hipStream_t stream) {
  (void)in_sizes; (void)n_in; (void)out_size; (void)ws_size;

  const float* q  = (const float*)d_in[0];
  const float* k  = (const float*)d_in[1];
  const float* v  = (const float*)d_in[2];
  const float* Wq = (const float*)d_in[3];
  const float* bq = (const float*)d_in[4];
  const float* Wk = (const float*)d_in[5];
  const float* bk = (const float*)d_in[6];
  const float* Wv = (const float*)d_in[7];
  const float* bv = (const float*)d_in[8];
  const float* Wo = (const float*)d_in[9];
  const float* bo = (const float*)d_in[10];
  float* out = (float*)d_out;

  const size_t BSD = (size_t)AB * ASEQ * ADM;  // 4 Mi elements
  bf16* Qb = (bf16*)d_ws;
  bf16* Kb = Qb + BSD;
  bf16* Vb = Kb + BSD;
  bf16* Ab = Vb + BSD;   // attention output (pre-Wo), bf16

  dim3 gemmGrid(GN / 64, GM / 128);   // (16, 32)
  dim3 blk(256);

  gemm_bias_kernel<float, bf16><<<gemmGrid, blk, 0, stream>>>(q, Wq, bq, Qb);
  gemm_bias_kernel<float, bf16><<<gemmGrid, blk, 0, stream>>>(k, Wk, bk, Kb);
  gemm_bias_kernel<float, bf16><<<gemmGrid, blk, 0, stream>>>(v, Wv, bv, Vb);

  dim3 attnGrid(ASEQ / 128, AH, AB);  // (16, 16, 2)
  mha_attention_kernel<<<attnGrid, blk, 0, stream>>>(Qb, Kb, Vb, Ab);

  gemm_bias_kernel<bf16, float><<<gemmGrid, blk, 0, stream>>>(Ab, Wo, bo, out);
}
